// SUGRL_Fast_59141699666065
// MI455X (gfx1250) — matrix operand order repaired and checked by reading, not compiled, not run on hardware
//
#include <hip/hip_runtime.h>
#include <stdint.h>

// ---------------------------------------------------------------------------
// MI455X (gfx1250) fused GCN-style forward:
//   h   = relu(A @ W1 + b1)        A:[N,512]  W1:[512,512]
//   h_a = h @ W2 + b2              W2:[512,128]
//   h_p = segment_sum(vals * h_a[cols], rows)   (rows sorted)
//
// GEMMs run on v_wmma_f32_16x16x32_bf16 using split-bf16 (hi/lo) emulation
// of f32 (3 WMMA products per tile, f32 accumulation). Both GEMMs are fused
// per 64-row block with h kept in LDS (saves ~410MB of HBM round-trip).
// Next-chunk weights/activations are prefetched (global_prefetch_b8) under
// the WMMA shadow. Aggregation uses register run-length accumulation over
// sorted rows with global_atomic_add_f32 flushes (h_a resident in 192MB L2).
// ---------------------------------------------------------------------------

typedef __attribute__((ext_vector_type(16))) __bf16 v16bf;
typedef __attribute__((ext_vector_type(8)))  __bf16 v8bf;
typedef __attribute__((ext_vector_type(8)))  float  v8f;

#define N_NODES 100000
#define N_EDGES 3200000
#define NIN 512
#define H1  512
#define H2  128
#define MTILE 64            // rows per block
#define KC 32               // K chunk per staging round (one bf16-WMMA K depth)

// LDS strides in 16-bit elements.
// WSTR=40 -> 80B row stride: 16B aligned, and lane stride 20 words is
// coprime-enough with 64 banks (L*20 mod 64 distinct for L=0..15) => no
// bank conflicts for half-wave B-operand reads.
#define WSTR 40
// HSTR=520 -> 1040B row stride: 16B aligned, lane bank stride 4 over 16 lanes
// spans all banks uniquely => conflict-free A-operand reads in phase B.
#define HSTR 520

#define SMEM_BYTES (2 * (MTILE*HSTR + MTILE*HSTR + H2*WSTR + H2*WSTR))  // 153600

__device__ __forceinline__ void splitbf(float x, unsigned short& hi, unsigned short& lo) {
    unsigned u = __float_as_uint(x);
    hi = (unsigned short)(u >> 16);                       // truncated bf16 high part
    float hf = __uint_as_float(((unsigned)hi) << 16);
    lo = (unsigned short)(__float_as_uint(x - hf) >> 16); // bf16 residual
}

__device__ __forceinline__ v16bf ld2x8(const unsigned short* p0, const unsigned short* p1) {
    v8bf a = *(const v8bf*)p0;   // ds_load_b128
    v8bf b = *(const v8bf*)p1;   // ds_load_b128
    return __builtin_shufflevector(a, b, 0,1,2,3,4,5,6,7,8,9,10,11,12,13,14,15);
}

#define WMMA_BF16(A, B, C) \
    __builtin_amdgcn_wmma_f32_16x16x32_bf16(false, (A), false, (B), (short)0, (C), false, false)

__global__ __launch_bounds__(256) void mlp_fused(
    const float* __restrict__ A, const float* __restrict__ W1,
    const float* __restrict__ b1, const float* __restrict__ W2,
    const float* __restrict__ b2, float* __restrict__ h_a, int nrows)
{
    extern __shared__ unsigned short smem[];
    // Phase-A views (weight/act staging), reused later:
    unsigned short* WtHi = smem;                  // [H1][WSTR] transposed W1 chunk (hi)
    unsigned short* WtLo = WtHi + H1 * WSTR;      // (lo)
    unsigned short* AHi  = WtLo + H1 * WSTR;      // [MTILE][WSTR] A chunk (hi)
    unsigned short* ALo  = AHi  + MTILE * WSTR;   // (lo)
    // Phase-B views (h buffer + W2 staging):
    unsigned short* hHi  = smem;                  // [MTILE][HSTR]
    unsigned short* hLo  = hHi  + MTILE * HSTR;
    unsigned short* W2tHi= hLo  + MTILE * HSTR;   // [H2][WSTR]
    unsigned short* W2tLo= W2tHi + H2 * WSTR;

    const int tid    = threadIdx.x;
    const int wave   = tid >> 5;
    const int lane   = tid & 31;
    const int lan16  = lane & 15;
    const int hiHalf = lane >> 4;                 // lanes 16..31
    const int rowBase = blockIdx.x * MTILE;

    const int rg = wave >> 1;                     // row-group 0..3 (16 rows each)
    const int nh = wave & 1;                      // N half (256 cols) in phase A

    const v8f vzero = {0.f,0.f,0.f,0.f,0.f,0.f,0.f,0.f};
    v8f acc[16];
#pragma unroll
    for (int t = 0; t < 16; ++t) acc[t] = vzero;

    // ------------------- Phase A: h = relu(A @ W1 + b1) --------------------
    for (int k0 = 0; k0 < NIN; k0 += KC) {
        // Stage A tile (MTILE x KC) split to bf16 hi/lo. One thread: 8 k's of one row.
        {
            int r  = tid >> 2;
            int kk = (tid & 3) * 8;
            int grow = rowBase + r;
            float x[8] = {0,0,0,0,0,0,0,0};
            if (grow < nrows) {
                const float4* src = (const float4*)(A + (size_t)grow * NIN + k0 + kk);
                float4 a0 = src[0], a1 = src[1];
                x[0]=a0.x; x[1]=a0.y; x[2]=a0.z; x[3]=a0.w;
                x[4]=a1.x; x[5]=a1.y; x[6]=a1.z; x[7]=a1.w;
            }
            unsigned hi[8], lo[8]; unsigned short h16, l16;
#pragma unroll
            for (int i = 0; i < 8; ++i) { splitbf(x[i], h16, l16); hi[i]=h16; lo[i]=l16; }
            uint4 ph = { hi[0]|(hi[1]<<16), hi[2]|(hi[3]<<16), hi[4]|(hi[5]<<16), hi[6]|(hi[7]<<16) };
            uint4 pl = { lo[0]|(lo[1]<<16), lo[2]|(lo[3]<<16), lo[4]|(lo[5]<<16), lo[6]|(lo[7]<<16) };
            *(uint4*)(AHi + r*WSTR + kk) = ph;
            *(uint4*)(ALo + r*WSTR + kk) = pl;
        }
        // Stage W1 chunk (KC x H1) TRANSPOSED into [n][k], split hi/lo.
        // One unit = 4 consecutive k at one n; lanes sweep n => coalesced loads.
#pragma unroll
        for (int i = 0; i < 16; ++i) {
            int u = i * 256 + tid;
            int n = u & (H1 - 1);
            int k = (u >> 9) * 4;
            unsigned hi[4], lo[4]; unsigned short h16, l16;
#pragma unroll
            for (int j = 0; j < 4; ++j) {
                float w = W1[(size_t)(k0 + k + j) * H1 + n];
                splitbf(w, h16, l16); hi[j]=h16; lo[j]=l16;
            }
            uint2 ph = { hi[0]|(hi[1]<<16), hi[2]|(hi[3]<<16) };
            uint2 pl = { lo[0]|(lo[1]<<16), lo[2]|(lo[3]<<16) };
            *(uint2*)(WtHi + n*WSTR + k) = ph;
            *(uint2*)(WtLo + n*WSTR + k) = pl;
        }
        __syncthreads();
        // Prefetch next K chunk into WGP$/L2 under the WMMA shadow
        // (global_prefetch_b8; no counter cost, no VGPR results).
        if (k0 + KC < NIN) {
            const char* nw = (const char*)(W1 + (size_t)(k0 + KC) * H1);
            __builtin_prefetch(nw + tid * 256, 0, 1);      // 256 x 256B = 64KB chunk
            if (tid < MTILE) {
                int grow = rowBase + tid;
                if (grow < nrows)
                    __builtin_prefetch(A + (size_t)grow * NIN + k0 + KC, 0, 1);
            }
        }
        // WMMA accumulate this K chunk. A-operand in contiguous k order (matches B):
        // lanes 0-15 take K[0..15], lanes 16-31 take K[16..31].
        {
            int koff = hiHalf ? 16 : 0;
            int arow = rg * 16 + lan16;
            const unsigned short* ap  = AHi + arow*WSTR;
            const unsigned short* alp = ALo + arow*WSTR;
            v16bf aHi = ld2x8(ap  + koff, ap  + 8 + koff);
            v16bf aLo = ld2x8(alp + koff, alp + 8 + koff);
            int ks = hiHalf ? 16 : 0;   // B: lane half selects K half
#pragma unroll
            for (int t = 0; t < 16; ++t) {
                int n = nh*256 + t*16 + lan16;
                const unsigned short* bp  = WtHi + n*WSTR + ks;
                const unsigned short* blp = WtLo + n*WSTR + ks;
                v16bf bHi = ld2x8(bp,  bp  + 8);
                v16bf bLo = ld2x8(blp, blp + 8);
                acc[t] = WMMA_BF16(aHi, bHi, acc[t]);
                acc[t] = WMMA_BF16(aHi, bLo, acc[t]);
                acc[t] = WMMA_BF16(aLo, bHi, acc[t]);
            }
        }
        __syncthreads();
    }
    // Bias + ReLU, split-store h tile into LDS (reusing staging space).
#pragma unroll
    for (int t = 0; t < 16; ++t) {
        int n = nh*256 + t*16 + lan16;
        float bias = b1[n];
        int mbase = rg*16 + hiHalf*8;           // C layout: M = r + 8*(lane>=16)
#pragma unroll
        for (int r = 0; r < 8; ++r) {
            float v = acc[t][r] + bias;
            v = v > 0.0f ? v : 0.0f;
            unsigned short h16, l16; splitbf(v, h16, l16);
            hHi[(mbase + r)*HSTR + n] = h16;
            hLo[(mbase + r)*HSTR + n] = l16;
        }
    }
    __syncthreads();

    // ------------------- Phase B: h_a = h @ W2 + b2 ------------------------
    v8f bacc[4];
#pragma unroll
    for (int t = 0; t < 4; ++t) bacc[t] = vzero;
    const int nb = nh * 64;                      // wave covers 64 of 128 out cols
    for (int k0 = 0; k0 < H1; k0 += KC) {
        // Stage W2 chunk (KC x H2) transposed, split hi/lo.
#pragma unroll
        for (int i = 0; i < 4; ++i) {
            int u = i * 256 + tid;
            int n = u & (H2 - 1);
            int k = (u >> 7) * 4;
            unsigned hi[4], lo[4]; unsigned short h16, l16;
#pragma unroll
            for (int j = 0; j < 4; ++j) {
                float w = W2[(size_t)(k0 + k + j) * H2 + n];
                splitbf(w, h16, l16); hi[j]=h16; lo[j]=l16;
            }
            uint2 ph = { hi[0]|(hi[1]<<16), hi[2]|(hi[3]<<16) };
            uint2 pl = { lo[0]|(lo[1]<<16), lo[2]|(lo[3]<<16) };
            *(uint2*)(W2tHi + n*WSTR + k) = ph;
            *(uint2*)(W2tLo + n*WSTR + k) = pl;
        }
        __syncthreads();
        // Prefetch next W2 chunk (16KB -> 64B/thread) under WMMA shadow.
        if (k0 + KC < H1) {
            const char* nw = (const char*)(W2 + (size_t)(k0 + KC) * H2);
            __builtin_prefetch(nw + tid * 64, 0, 1);
        }
        int koff = hiHalf ? 16 : 0;
        int arow = rg*16 + lan16;
        const unsigned short* hp = hHi + arow*HSTR + k0;
        const unsigned short* lp = hLo + arow*HSTR + k0;
        v16bf aHi = ld2x8(hp + koff, hp + 8 + koff);
        v16bf aLo = ld2x8(lp + koff, lp + 8 + koff);
        int ks = hiHalf ? 16 : 0;
#pragma unroll
        for (int t = 0; t < 4; ++t) {
            int n = nb + t*16 + lan16;
            const unsigned short* bp  = W2tHi + n*WSTR + ks;
            const unsigned short* blp = W2tLo + n*WSTR + ks;
            v16bf bHi = ld2x8(bp,  bp  + 8);
            v16bf bLo = ld2x8(blp, blp + 8);
            bacc[t] = WMMA_BF16(aHi, bHi, bacc[t]);
            bacc[t] = WMMA_BF16(aHi, bLo, bacc[t]);
            bacc[t] = WMMA_BF16(aLo, bHi, bacc[t]);
        }
        __syncthreads();
    }
    // Write h_a (first half of d_out).
#pragma unroll
    for (int t = 0; t < 4; ++t) {
        int n = nb + t*16 + lan16;
        float bias = b2[n];
        int mbase = rowBase + rg*16 + hiHalf*8;
#pragma unroll
        for (int r = 0; r < 8; ++r) {
            int m = mbase + r;
            if (m < nrows) h_a[(size_t)m*H2 + n] = bacc[t][r] + bias;
        }
    }
}

// --------------------- Sorted-COO aggregation --------------------------------
// One wave owns 32 consecutive edges. Lanes tile the 128-wide feature (float4
// each). Accumulate in registers while the (sorted) row id is unchanged; flush
// via global_atomic_add_f32 only at run boundaries. h_a stays resident in L2.
#define EPW 32
__global__ __launch_bounds__(256) void edge_agg(
    const float* __restrict__ h_a, const int* __restrict__ rows,
    const int* __restrict__ cols, const float* __restrict__ vals,
    float* __restrict__ h_p, int nedges)
{
    int wave = blockIdx.x * (blockDim.x >> 5) + (threadIdx.x >> 5);
    int lane = threadIdx.x & 31;
    long base = (long)wave * EPW;
    if (base >= nedges) return;

    float ax = 0.f, ay = 0.f, az = 0.f, aw = 0.f;
#pragma unroll 4
    for (int i = 0; i < EPW; ++i) {
        long e = base + i;
        if (e >= nedges) break;
        int   c = cols[e];
        float v = vals[e];
        float4 x = ((const float4*)(h_a + (size_t)c * H2))[lane];
        ax += v * x.x; ay += v * x.y; az += v * x.z; aw += v * x.w;
        int r = rows[e];
        bool flush = (i == EPW - 1) || (e + 1 >= nedges) || (rows[e + 1] != r);
        if (flush) {
            float* dst = h_p + (size_t)r * H2 + lane * 4;
            atomicAdd(dst + 0, ax);
            atomicAdd(dst + 1, ay);
            atomicAdd(dst + 2, az);
            atomicAdd(dst + 3, aw);
            ax = ay = az = aw = 0.f;
        }
    }
}

extern "C" void kernel_launch(void* const* d_in, const int* in_sizes, int n_in,
                              void* d_out, int out_size, void* d_ws, size_t ws_size,
                              hipStream_t stream) {
    (void)in_sizes; (void)n_in; (void)out_size; (void)d_ws; (void)ws_size;
    const float* seq_a = (const float*)d_in[0];
    const float* W1    = (const float*)d_in[1];
    const float* b1    = (const float*)d_in[2];
    const float* W2    = (const float*)d_in[3];
    const float* b2    = (const float*)d_in[4];
    const int*   rows  = (const int*)d_in[5];
    const int*   cols  = (const int*)d_in[6];
    const float* vals  = (const float*)d_in[7];

    float* h_a = (float*)d_out;                       // [N, 128]
    float* h_p = h_a + (size_t)N_NODES * H2;          // [N, 128]

    // Allow >64KB dynamic LDS (150KB; 2 blocks/WGP out of 320KB).
    hipFuncSetAttribute(reinterpret_cast<const void*>(mlp_fused),
                        hipFuncAttributeMaxDynamicSharedMemorySize, SMEM_BYTES);

    // Zero h_p (atomic accumulation target); stream-ordered, capture-safe.
    hipMemsetAsync(h_p, 0, (size_t)N_NODES * H2 * sizeof(float), stream);

    int grid1 = (N_NODES + MTILE - 1) / MTILE;        // 1563 row tiles
    mlp_fused<<<grid1, 256, SMEM_BYTES, stream>>>(seq_a, W1, b1, W2, b2, h_a, N_NODES);

    int nwaves = (N_EDGES + EPW - 1) / EPW;           // 100000 edge waves
    int grid2  = (nwaves + 7) / 8;                    // 8 waves per block
    edge_agg<<<grid2, 256, 0, stream>>>(h_a, rows, cols, vals, h_p, N_EDGES);
}
